// DAGNN_54176717472059
// MI455X (gfx1250) — matrix operand order repaired and checked by reading, not compiled, not run on hardware
//
#include <hip/hip_runtime.h>
#include <math.h>

// ---------------------------------------------------------------------------
// DAG-NN propagation for MI455X (gfx1250, wave32, WMMA).
//   - per-wave private 16-row z tile in LDS (f16, padded pitch -> no bank
//     conflicts); waves fully independent: zero barriers
//   - hidden scan in 32-node blocks: K loop is branch-free, every chunk full,
//     one A fragment feeds 2 v_wmma_f32_16x16x32_f16
//   - W pre-converted once into a B-fragment-tiled f16 layout in d_ws
//     (coalesced 1KB blocks, no per-iteration cvt); fp32 fallback if ws small
//   - 32x32 strictly-lower-triangular in-block recurrence in fp32 with
//     __shfl(width=16) column broadcasts + hardware tanh
// ---------------------------------------------------------------------------

typedef __attribute__((ext_vector_type(16))) _Float16 v16h;
typedef __attribute__((ext_vector_type(8)))  _Float16 v8h;
typedef __attribute__((ext_vector_type(4)))  _Float16 v4h;
typedef __attribute__((ext_vector_type(8)))  float    v8f;
typedef __attribute__((ext_vector_type(4)))  float    v4f;

#define NN        1024   // total nodes
#define NIN       256    // input nodes
#define NOUT      256    // output nodes
#define NHID_END  768    // NN - NOUT
#define BATCH     8192
#define ZP        776    // LDS z row pitch in halves (768 + 8 pad)
#define WPW       2      // waves per workgroup
#define RPW       16     // batch rows per wave
#define KTILES    (NN / 32)   // 32 k-chunks per node-tile row

__device__ __forceinline__ float fast_tanh(float x) {
#if __has_builtin(__builtin_amdgcn_tanhf)
  return __builtin_amdgcn_tanhf(x);      // v_tanh_f32 on gfx1250
#else
  return tanhf(x);
#endif
}

__device__ __forceinline__ v8f wmma_f16(v16h a, v16h bf, v8f c) {
  return __builtin_amdgcn_wmma_f32_16x16x32_f16(false, a, false, bf,
                                                (short)0, c, false, false);
}

// A fragment (16x32 f16, M=batch rows, K=past nodes) from this wave's LDS z
// tile. Lanes 0-15: M=lane, K {k0..+15}; lanes 16-31: M=lane-16,
// K {k0+16..+31}.  All chunks full -> branch-free.
__device__ __forceinline__ v16h load_a(const _Float16* zrow, int k0, int kofs) {
  v8h lo  = *(const v8h*)(zrow + k0 + 2 * kofs);
  v8h hiv = *(const v8h*)(zrow + k0 + 2 * kofs + 8);
  v16h a;
#pragma unroll
  for (int i = 0; i < 8; ++i) { a[i] = lo[i]; a[i + 8] = hiv[i]; }
  return a;
}

// B fragment from the pre-tiled f16 W: tile (nodeBase/16, k0/32) is 1KB
// contiguous, lane's 16 halves at lane*16 -> two coalesced global_load_b128.
__device__ __forceinline__ v16h load_b_tiled(const _Float16* __restrict__ Wh,
                                             int nodeBase, int k0, int lane) {
  const _Float16* p =
      Wh + ((size_t)((nodeBase >> 4) * KTILES + (k0 >> 5)) << 9) + lane * 16;
  v8h lo  = *(const v8h*)p;
  v8h hiv = *(const v8h*)(p + 8);
  v16h r;
#pragma unroll
  for (int i = 0; i < 8; ++i) { r[i] = lo[i]; r[i + 8] = hiv[i]; }
  return r;
}

// Fallback: B fragment straight from fp32 W with on-the-fly conversion.
__device__ __forceinline__ v16h load_b_f32(const float* __restrict__ W,
                                           int nodeBase, int n, int kb) {
  const float* wr = W + (size_t)(nodeBase + n) * NN + kb;
  v4f w0 = *(const v4f*)(wr + 0), w1 = *(const v4f*)(wr + 4);
  v4f w2 = *(const v4f*)(wr + 8), w3 = *(const v4f*)(wr + 12);
  v16h bf;
#pragma unroll
  for (int i = 0; i < 4; ++i) {
    bf[i] = (_Float16)w0[i];      bf[i + 4] = (_Float16)w1[i];
    bf[i + 8] = (_Float16)w2[i];  bf[i + 12] = (_Float16)w3[i];
  }
  return bf;
}

// One-shot: convert W (fp32 row-major) into B-fragment-tiled f16 layout.
// One wave per (tileRow, kChunk); each lane emits its own 32B segment.
__global__ __launch_bounds__(256) void convw_kernel(
    const float* __restrict__ W, _Float16* __restrict__ Wh) {
  const int lane = threadIdx.x & 31;
  const int tile = blockIdx.x * 8 + (threadIdx.x >> 5);  // 0 .. 2047
  const int tr = tile / KTILES;                          // node tile row
  const int kc = tile % KTILES;                          // k chunk
  const int n = lane & 15, hi = lane >> 4;
  const float* src = W + (size_t)(tr * 16 + n) * NN + kc * 32 + hi * 16;
  v4f s0 = *(const v4f*)(src + 0), s1 = *(const v4f*)(src + 4);
  v4f s2 = *(const v4f*)(src + 8), s3 = *(const v4f*)(src + 12);
  v8h lo, hv;
#pragma unroll
  for (int i = 0; i < 4; ++i) {
    lo[i] = (_Float16)s0[i]; lo[i + 4] = (_Float16)s1[i];
    hv[i] = (_Float16)s2[i]; hv[i + 4] = (_Float16)s3[i];
  }
  _Float16* dst = Wh + ((size_t)tile << 9) + lane * 16;
  *(v8h*)dst = lo;
  *(v8h*)(dst + 8) = hv;
}

template <bool TILED>
__global__ __launch_bounds__(WPW * 32) void dagnn_kernel(
    const float* __restrict__ x,         // [BATCH, NIN]
    const float* __restrict__ W,         // [NN, NN] fp32
    const _Float16* __restrict__ Wh,     // tiled f16 W (TILED path)
    const float* __restrict__ b,         // [NN]
    float* __restrict__ out) {           // [BATCH, NOUT]
  __shared__ _Float16 zsh[WPW * RPW * ZP];   // 48.5 KB

  const int lane    = threadIdx.x & 31;
  const int wave    = threadIdx.x >> 5;
  const int rowBase = (blockIdx.x * WPW + wave) * RPW;
  _Float16* zw = zsh + wave * RPW * ZP;      // wave-private 16 x ZP tile

  const int n    = lane & 15;
  const int hi   = lane >> 4;
  const int kofs = hi ? 8 : 0;
  const _Float16* zrow = zw + n * ZP;

  // ---- stage inputs: z[m][0..255] = f16(x[rowBase+m][:]) ----
  for (int g = lane; g < RPW * (NIN / 4); g += 32) {
    int m = g >> 6, c4 = g & 63;
    v4f xv = *(const v4f*)(x + (size_t)(rowBase + m) * NIN + c4 * 4);
    v4h hv;
    hv[0] = (_Float16)xv[0]; hv[1] = (_Float16)xv[1];
    hv[2] = (_Float16)xv[2]; hv[3] = (_Float16)xv[3];
    *(v4h*)(zw + m * ZP + c4 * 4) = hv;
  }

  // ---- sequential hidden scan: 16 blocks of 32 nodes ----
  for (int blk = 0; blk < (NHID_END - NIN) / 32; ++blk) {
    const int i0 = NIN + blk * 32;       // multiple of 32 -> all chunks full

    v8f c0 = {}, c1 = {};                // nodes i0+n / i0+16+n, rows v(+8)
    for (int k0 = 0; k0 < i0; k0 += 32) {
      v16h a  = load_a(zrow, k0, kofs);
      v16h b0 = TILED ? load_b_tiled(Wh, i0, k0, lane)
                      : load_b_f32(W, i0, n, k0 + hi * 16);
      v16h b1 = TILED ? load_b_tiled(Wh, i0 + 16, k0, lane)
                      : load_b_f32(W, i0 + 16, n, k0 + hi * 16);
      c0 = wmma_f16(a, b0, c0);
      c1 = wmma_f16(a, b1, c1);
    }
    const float bi0 = b[i0 + n], bi1 = b[i0 + 16 + n];
#pragma unroll
    for (int v = 0; v < 8; ++v) { c0[v] += bi0; c1[v] += bi1; }

    // in-block 32x32 strictly-lower-triangular recurrence (fp32 exact)
    float W0[16], W1[32];                // W[i0+n][i0..], W[i0+16+n][i0..]
    {
      const float* r0 = W + (size_t)(i0 + n) * NN + i0;
      const float* r1 = W + (size_t)(i0 + 16 + n) * NN + i0;
#pragma unroll
      for (int q = 0; q < 4; ++q) {
        v4f a0 = *(const v4f*)(r0 + q * 4);
        v4f a1 = *(const v4f*)(r1 + q * 4);
        v4f a2 = *(const v4f*)(r1 + 16 + q * 4);
#pragma unroll
        for (int i = 0; i < 4; ++i) {
          W0[q * 4 + i] = a0[i];
          W1[q * 4 + i] = a1[i];
          W1[16 + q * 4 + i] = a2[i];
        }
      }
    }
    // phase 1: t = 0..15 (z source = c0)
#pragma unroll
    for (int t = 0; t < 16; ++t) {
#pragma unroll
      for (int v = 0; v < 8; ++v) {
        float a_t = __shfl(c0[v], t, 16);
        float z_t = fast_tanh(a_t);
        if (n > t) c0[v] += W0[t] * z_t;
        c1[v] += W1[t] * z_t;            // n+16 > t always (t < 16)
        if (n == t) zw[(v + (hi << 3)) * ZP + i0 + t] = (_Float16)z_t;
      }
    }
    // phase 2: t = 16..31 (z source = c1)
#pragma unroll
    for (int t2 = 0; t2 < 16; ++t2) {
#pragma unroll
      for (int v = 0; v < 8; ++v) {
        float a_t = __shfl(c1[v], t2, 16);
        float z_t = fast_tanh(a_t);
        if (n > t2) c1[v] += W1[16 + t2] * z_t;
        if (n == t2) zw[(v + (hi << 3)) * ZP + i0 + 16 + t2] = (_Float16)z_t;
      }
    }
  }

  // ---- output GEMM: Y = Z[:, :768] @ W[768:1024, :768]^T ----
  // 4 groups of 4 out-tiles: one A fragment feeds 4 WMMAs per chunk.
#pragma unroll 1
  for (int ng = 0; ng < NOUT / 64; ++ng) {
    v8f acc0 = {}, acc1 = {}, acc2 = {}, acc3 = {};
    for (int k0 = 0; k0 < NHID_END; k0 += 32) {
      v16h a = load_a(zrow, k0, kofs);
      const int nb = NHID_END + ng * 64;
      v16h f0 = TILED ? load_b_tiled(Wh, nb + 0,  k0, lane)
                      : load_b_f32(W, nb + 0,  n, k0 + hi * 16);
      v16h f1 = TILED ? load_b_tiled(Wh, nb + 16, k0, lane)
                      : load_b_f32(W, nb + 16, n, k0 + hi * 16);
      v16h f2 = TILED ? load_b_tiled(Wh, nb + 32, k0, lane)
                      : load_b_f32(W, nb + 32, n, k0 + hi * 16);
      v16h f3 = TILED ? load_b_tiled(Wh, nb + 48, k0, lane)
                      : load_b_f32(W, nb + 48, n, k0 + hi * 16);
      acc0 = wmma_f16(a, f0, acc0);
      acc1 = wmma_f16(a, f1, acc1);
      acc2 = wmma_f16(a, f2, acc2);
      acc3 = wmma_f16(a, f3, acc3);
    }
#pragma unroll
    for (int v = 0; v < 8; ++v) {
      float* orow = out + (size_t)(rowBase + v + (hi << 3)) * NOUT + ng * 64 + n;
      orow[0]  = acc0[v];
      orow[16] = acc1[v];
      orow[32] = acc2[v];
      orow[48] = acc3[v];
    }
  }
}

extern "C" void kernel_launch(void* const* d_in, const int* in_sizes, int n_in,
                              void* d_out, int out_size, void* d_ws,
                              size_t ws_size, hipStream_t stream) {
  (void)in_sizes; (void)n_in; (void)out_size;
  const float* x = (const float*)d_in[0];
  const float* W = (const float*)d_in[1];
  const float* b = (const float*)d_in[2];
  float* out = (float*)d_out;

  dim3 grid(BATCH / (WPW * RPW));  // 256 workgroups
  dim3 block(WPW * 32);            // 2 waves of 32

  const size_t whBytes = (size_t)NN * NN * sizeof(_Float16);  // 2 MB
  if (d_ws != nullptr && ws_size >= whBytes) {
    _Float16* Wh = (_Float16*)d_ws;
    convw_kernel<<<dim3((NN / 16) * KTILES / 8), dim3(256), 0, stream>>>(W, Wh);
    dagnn_kernel<true><<<grid, block, 0, stream>>>(x, W, Wh, b, out);
  } else {
    dagnn_kernel<false><<<grid, block, 0, stream>>>(x, W, nullptr, b, out);
  }
}
